// StereoAwareEncoder_39651138076865
// MI455X (gfx1250) — compile-verified, matrix operand order checked by reading, not run on hardware
//
#include <hip/hip_runtime.h>
#include <hip/hip_bf16.h>
#include <math.h>

#define N_NODES 100000
#define N_EDGES 400000
#define N_GRAPHS 4096
#define F_IN 21
#define HID 128
#define NHEAD 4
#define CH 32
#define NLAYER 4
#define KPAD_IN 32

typedef float v8f __attribute__((ext_vector_type(8)));
typedef __attribute__((ext_vector_type(16))) __bf16 v16bf;
typedef unsigned int u32x4 __attribute__((ext_vector_type(4)));
typedef int i32x8 __attribute__((ext_vector_type(8)));
typedef int i32x4 __attribute__((ext_vector_type(4)));

union FragBF {
    uint4 q[2];
    v16bf v;
};

__device__ __forceinline__ unsigned short f2bf(float f) {
    unsigned int u = __float_as_uint(f);
    u += 0x7FFFu + ((u >> 16) & 1u);   // round to nearest even
    return (unsigned short)(u >> 16);
}

__device__ __forceinline__ void atomicMaxF(float* addr, float val) {
    // IEEE ordering trick: works with -inf init, mixed signs
    if (val >= 0.0f) atomicMax((int*)addr, __float_as_int(val));
    else             atomicMin((unsigned int*)addr, __float_as_uint(val));
}

// ---- Tensor Data Mover: 1-D global->LDS DMA (data_size = 4B, single row tile)
// D# per CDNA5 ISA. nbytes/4 must be <= 65535.
__device__ __forceinline__ void tdm_load_bytes(unsigned int lds_off,
                                               const void* src,
                                               unsigned int nbytes) {
    unsigned long long ga = (unsigned long long)src;
    unsigned int nelem = nbytes >> 2;
    u32x4 g0;
    g0.x = 1u;                                      // count=1, user mode
    g0.y = lds_off;                                 // lds_addr [63:32]
    g0.z = (unsigned int)ga;                        // global_addr lo
    g0.w = (unsigned int)(ga >> 32) | (2u << 30);   // global_addr hi | type=2
    i32x8 g1;
    g1[0] = 0x00020000;                             // data_size=2 (4 bytes)
    g1[1] = (int)((nelem & 0xFFFFu) << 16);         // tensor_dim0[15:0] @ [63:48]
    g1[2] = (int)((nelem >> 16) | (1u << 16));      // tensor_dim0 hi | tensor_dim1=1 lo
    g1[3] = (int)(nelem << 16);                     // tensor_dim1 hi=0 | tile_dim0
    g1[4] = 1;                                      // tile_dim1=1, tile_dim2=0
    g1[5] = (int)nelem;                             // tensor_dim0_stride lo
    g1[6] = 0;                                      // stride hi, dim1_stride lo
    g1[7] = 0;
    i32x4 z4 = {0, 0, 0, 0};
    i32x8 z8 = {0, 0, 0, 0, 0, 0, 0, 0};
    __builtin_amdgcn_tensor_load_to_lds(g0, g1, z4, z4, z8, 0);
}

// ---------------------------------------------------------------- conversions
__global__ void k_cvt_x(const float* __restrict__ x, unsigned short* __restrict__ xb) {
    int idx = blockIdx.x * blockDim.x + threadIdx.x;
    if (idx >= N_NODES * KPAD_IN) return;
    int node = idx >> 5, kk = idx & 31;
    xb[idx] = (kk < F_IN) ? f2bf(x[(size_t)node * F_IN + kk]) : (unsigned short)0;
}

// W[K,N] row-major  ->  Wt[N,Kp] bf16 (transposed, K padded with zeros)
__global__ void k_cvt_wt(const float* __restrict__ W, unsigned short* __restrict__ Wt,
                         int K, int N, int Kp) {
    int idx = blockIdx.x * blockDim.x + threadIdx.x;
    if (idx >= N * Kp) return;
    int n = idx / Kp, kk = idx % Kp;
    Wt[idx] = (kk < K) ? f2bf(W[(size_t)kk * N + n]) : (unsigned short)0;
}

__global__ void k_fill(float* __restrict__ p, float v, int n) {
    int idx = blockIdx.x * blockDim.x + threadIdx.x;
    if (idx < n) p[idx] = v;
}

// ---------------------------------------------------------------- WMMA GEMM
// C[M,128] = A_bf16[M,KP] * Bt_bf16[128,KP]^T + bias[128]
// Bt is the weight stored transposed (row n holds column n of W), so loading a
// B fragment is identical to loading an A fragment (ISA 16-bit A layout).
// Weight panel staged into LDS by the Tensor Data Mover (wave 0), then read
// back as ds_load_b128 fragments.
template <int KP>
__global__ __launch_bounds__(128) void k_gemm_bf16(
    const unsigned short* __restrict__ A,
    const unsigned short* __restrict__ Bt,
    const float* __restrict__ bias,
    float* __restrict__ C, int M)
{
    __shared__ unsigned short sB[HID * KP];
    const int t = threadIdx.x;
    if (t < 32) {
        tdm_load_bytes((unsigned int)(unsigned long long)(&sB[0]), Bt,
                       HID * KP * 2);
        __builtin_amdgcn_s_wait_tensorcnt(0);
    }
    __syncthreads();

    const int wave = t >> 5, lane = t & 31;
    const int l  = lane & 15;                       // row/col within tile
    const int hi = lane >> 4;                       // K-half selector
    const int m0 = blockIdx.x * 64 + wave * 16;
    if (m0 >= M) return;                            // wave-uniform

    constexpr int NK = KP >> 5;                     // k-steps of 32

    FragBF a[NK];
    const uint4* arow = reinterpret_cast<const uint4*>(A + (size_t)(m0 + l) * KP);
#pragma unroll
    for (int ks = 0; ks < NK; ++ks) {
        const int q0 = ks * 4 + hi;                 // 32 ushorts = 4 uint4 per step
        a[ks].q[0] = arow[q0];                      // K {0..7}+off
        a[ks].q[1] = arow[q0 + 2];                  // K {16..23}+off
    }

#pragma unroll
    for (int nt = 0; nt < 8; ++nt) {
        const int n0 = nt * 16;
        v8f acc = {};
        const uint4* brow = reinterpret_cast<const uint4*>(sB + (size_t)(n0 + l) * KP);
#pragma unroll
        for (int ks = 0; ks < NK; ++ks) {
            FragBF b;
            const int q0 = ks * 4 + hi;
            b.q[0] = brow[q0];
            b.q[1] = brow[q0 + 2];
            acc = __builtin_amdgcn_wmma_f32_16x16x32_bf16(
                      false, a[ks].v, false, b.v, (short)0, acc, false, false);
        }
        // D layout: VGPR j -> row m0 + hi*8 + j, col n0 + l
        const float bb = bias[n0 + l];
        float* crow = C + (size_t)(m0 + hi * 8) * HID + n0 + l;
#pragma unroll
        for (int j = 0; j < 8; ++j) crow[(size_t)j * HID] = acc[j] + bb;
    }
}

// ---------------------------------------------------------------- batchnorm
__global__ void k_bn_stats(const float* __restrict__ X, const float* __restrict__ addv,
                           float* __restrict__ stats, int M) {
    const int c  = threadIdx.x & 127;
    const int r0 = blockIdx.x * 2 + (threadIdx.x >> 7);
    const int stride = gridDim.x * 2;
    const float add = addv ? addv[c] : 0.0f;
    float s = 0.0f, s2 = 0.0f;
    for (int r = r0; r < M; r += stride) {
        float v = X[(size_t)r * HID + c] + add;
        s += v; s2 += v * v;
    }
    atomicAdd(&stats[c], s);
    atomicAdd(&stats[HID + c], s2);
}

// flags: bit0 = relu, bit1 = residual add (applied after relu)
__global__ void k_bn_apply(const float* __restrict__ X, const float* __restrict__ addv,
                           const float* __restrict__ stats,
                           const float* __restrict__ g, const float* __restrict__ b,
                           const float* __restrict__ resid,
                           float* __restrict__ hout, unsigned short* __restrict__ hbo,
                           int total, float invM, int flags) {
    int idx = blockIdx.x * blockDim.x + threadIdx.x;
    if (idx >= total) return;
    int c = idx & 127;
    float mu  = stats[c] * invM;
    float var = stats[HID + c] * invM - mu * mu;
    float v = X[idx] + (addv ? addv[c] : 0.0f);
    float y = (v - mu) * rsqrtf(var + 1e-5f) * g[c] + b[c];
    if (flags & 1) y = fmaxf(y, 0.0f);
    if (flags & 2) y += resid[idx];
    hout[idx] = y;
    if (hbo) hbo[idx] = f2bf(y);
}

// ---------------------------------------------------------------- GAT edges
__global__ void k_gat_logit(const float* __restrict__ xl, const float* __restrict__ xr,
                            const int* __restrict__ ei, const float* __restrict__ att,
                            float* __restrict__ logit, float* __restrict__ nmax, int total) {
    int idx = blockIdx.x * blockDim.x + threadIdx.x;
    if (idx >= total) return;
    int e = idx >> 2, hh = idx & 3;
    int s, d;
    if (e < N_EDGES) { s = ei[e]; d = ei[N_EDGES + e]; } else { s = d = e - N_EDGES; }
    const float* pl = xl + (size_t)s * HID + hh * CH;
    const float* pr = xr + (size_t)d * HID + hh * CH;
    const float* pa = att + hh * CH;
    float acc = 0.0f;
#pragma unroll
    for (int c = 0; c < CH; ++c) {
        float m = pl[c] + pr[c];
        m = (m > 0.0f) ? m : m * 0.2f;              // leaky relu
        acc += m * pa[c];
    }
    logit[idx] = acc;
    atomicMaxF(&nmax[d * NHEAD + hh], acc);
}

__global__ void k_expsum(float* __restrict__ logit, const int* __restrict__ ei,
                         const float* __restrict__ nmax, float* __restrict__ nsum,
                         int total, int withSelf) {
    int idx = blockIdx.x * blockDim.x + threadIdx.x;
    if (idx >= total) return;
    int e = idx >> 2, hh = idx & 3;
    int d = (withSelf && e >= N_EDGES) ? (e - N_EDGES) : ei[N_EDGES + e];
    float v = __expf(logit[idx] - nmax[d * NHEAD + hh]);
    logit[idx] = v;
    atomicAdd(&nsum[d * NHEAD + hh], v);
}

// out[dst,:] += (e / sum) * val[src,:]
__global__ void k_scatter(const float* __restrict__ val, const float* __restrict__ elog,
                          const float* __restrict__ nsum, const int* __restrict__ ei,
                          float* __restrict__ out, int total, int withSelf) {
    int idx = blockIdx.x * blockDim.x + threadIdx.x;
    if (idx >= total) return;
    int e = idx >> 7, f = idx & 127, hh = f >> 5;
    int s, d;
    if (withSelf && e >= N_EDGES) { s = d = e - N_EDGES; }
    else                          { s = ei[e]; d = ei[N_EDGES + e]; }
    float alpha = elog[e * NHEAD + hh] / (nsum[d * NHEAD + hh] + 1e-16f);
    atomicAdd(&out[(size_t)d * HID + f], alpha * val[(size_t)s * HID + f]);
}

// ---------------------------------------------------------------- transformer edges
__global__ void k_tr_logit(const float* __restrict__ q, const float* __restrict__ k,
                           const int* __restrict__ ei,
                           float* __restrict__ logit, float* __restrict__ nmax, int total) {
    int idx = blockIdx.x * blockDim.x + threadIdx.x;
    if (idx >= total) return;
    int e = idx >> 2, hh = idx & 3;
    int s = ei[e], d = ei[N_EDGES + e];
    const float* pq = q + (size_t)d * HID + hh * CH;
    const float* pk = k + (size_t)s * HID + hh * CH;
    float acc = 0.0f;
#pragma unroll
    for (int c = 0; c < CH; ++c) acc += pq[c] * pk[c];
    acc *= 0.17677669529663687f;                    // 1/sqrt(32)
    logit[idx] = acc;
    atomicMaxF(&nmax[d * NHEAD + hh], acc);
}

// ---------------------------------------------------------------- pooling
__global__ void k_count(const int* __restrict__ batch, float* __restrict__ cnt) {
    int i = blockIdx.x * blockDim.x + threadIdx.x;
    if (i < N_NODES) atomicAdd(&cnt[batch[i]], 1.0f);
}

__global__ void k_pool_acc(const float* __restrict__ h, const int* __restrict__ batch,
                           float* __restrict__ psum, float* __restrict__ pmax) {
    int idx = blockIdx.x * blockDim.x + threadIdx.x;
    if (idx >= N_NODES * HID) return;
    int node = idx >> 7, f = idx & 127;
    int g = batch[node];
    float v = h[idx];
    atomicAdd(&psum[(size_t)g * HID + f], v);
    atomicMaxF(&pmax[(size_t)g * HID + f], v);
}

__global__ void k_pool_final(const float* __restrict__ psum, const float* __restrict__ pmax,
                             const float* __restrict__ cnt, float* __restrict__ out) {
    int idx = blockIdx.x * blockDim.x + threadIdx.x;
    if (idx >= N_GRAPHS * HID) return;
    int g = idx >> 7, f = idx & 127;
    float c = fmaxf(cnt[g], 1.0f);
    out[(size_t)g * 2 * HID + f] = psum[idx] / c;
    float m = pmax[idx];
    out[(size_t)g * 2 * HID + HID + f] =
        (m >= -3.402823e38f && m <= 3.402823e38f) ? m : 0.0f;
}

// ---------------------------------------------------------------- host
extern "C" void kernel_launch(void* const* d_in, const int* in_sizes, int n_in,
                              void* d_out, int out_size, void* d_ws, size_t ws_size,
                              hipStream_t stream) {
    (void)in_sizes; (void)n_in; (void)out_size; (void)ws_size;
    // setup_inputs() dict order: x, edge_index, batch, then params in insertion order
    const float* x      = (const float*)d_in[0];
    const int*   ei     = (const int*)  d_in[1];
    const int*   batch  = (const int*)  d_in[2];
    const float* W_in   = (const float*)d_in[3];
    const float* b_in   = (const float*)d_in[4];
    const float* g_in   = (const float*)d_in[5];
    const float* be_in  = (const float*)d_in[6];
    const float* Wl     = (const float*)d_in[7];
    const float* Wr     = (const float*)d_in[8];
    const float* bl     = (const float*)d_in[9];
    const float* br     = (const float*)d_in[10];
    const float* att    = (const float*)d_in[11];
    const float* gbias  = (const float*)d_in[12];
    const float* g_gat  = (const float*)d_in[13];
    const float* be_gat = (const float*)d_in[14];
    const float* Wq     = (const float*)d_in[15];
    const float* bq     = (const float*)d_in[16];
    const float* Wk     = (const float*)d_in[17];
    const float* bk     = (const float*)d_in[18];
    const float* Wv     = (const float*)d_in[19];
    const float* bv     = (const float*)d_in[20];
    const float* Wskip  = (const float*)d_in[21];
    const float* bskip  = (const float*)d_in[22];
    const float* g_tr   = (const float*)d_in[23];
    const float* be_tr  = (const float*)d_in[24];
    float* out = (float*)d_out;

    char* cur = (char*)d_ws;
    auto alloc = [&](size_t b) -> void* {
        void* p = (void*)cur;
        cur += (b + 255) & ~(size_t)255;
        return p;
    };
    const size_t NM = (size_t)N_NODES * HID;
    unsigned short* xb   = (unsigned short*)alloc((size_t)N_NODES * KPAD_IN * 2);
    unsigned short* Wint = (unsigned short*)alloc((size_t)HID * KPAD_IN * 2);
    unsigned short* Wlt  = (unsigned short*)alloc((size_t)NLAYER * HID * HID * 2);
    unsigned short* Wrt  = (unsigned short*)alloc((size_t)NLAYER * HID * HID * 2);
    unsigned short* Wqt  = (unsigned short*)alloc((size_t)HID * HID * 2);
    unsigned short* Wkt  = (unsigned short*)alloc((size_t)HID * HID * 2);
    unsigned short* Wvt  = (unsigned short*)alloc((size_t)HID * HID * 2);
    unsigned short* Wst  = (unsigned short*)alloc((size_t)HID * HID * 2);
    unsigned short* hb   = (unsigned short*)alloc(NM * 2);
    float* h    = (float*)alloc(NM * 4);
    float* bufA = (float*)alloc(NM * 4);
    float* bufB = (float*)alloc(NM * 4);
    float* bufC = (float*)alloc(NM * 4);
    float* bufD = (float*)alloc(NM * 4);
    float* elog = (float*)alloc((size_t)(N_EDGES + N_NODES) * NHEAD * 4);
    float* nmax = (float*)alloc((size_t)N_NODES * NHEAD * 4);
    float* nsum = (float*)alloc((size_t)N_NODES * NHEAD * 4);
    float* stats = (float*)alloc(2 * HID * 4);
    float* psum = (float*)alloc((size_t)N_GRAPHS * HID * 4);
    float* pmax = (float*)alloc((size_t)N_GRAPHS * HID * 4);
    float* cnt  = (float*)alloc((size_t)N_GRAPHS * 4);

    auto fill = [&](float* p, float v, int n) {
        k_fill<<<dim3((n + 255) / 256), dim3(256), 0, stream>>>(p, v, n);
    };
    auto gemm128 = [&](const unsigned short* A, const unsigned short* Bt,
                       const float* bias, float* C) {
        k_gemm_bf16<HID><<<dim3((N_NODES + 63) / 64), dim3(128), 0, stream>>>(
            A, Bt, bias, C, N_NODES);
    };
    auto bn = [&](const float* X, const float* addv, const float* g, const float* b,
                  const float* resid, float* hout, unsigned short* hbo, int flags) {
        fill(stats, 0.0f, 2 * HID);
        k_bn_stats<<<dim3(512), dim3(256), 0, stream>>>(X, addv, stats, N_NODES);
        int tot = N_NODES * HID;
        k_bn_apply<<<dim3((tot + 255) / 256), dim3(256), 0, stream>>>(
            X, addv, stats, g, b, resid, hout, hbo, tot, 1.0f / N_NODES, flags);
    };

    // --- convert inputs/weights to bf16 (transposed weights) ---
    k_cvt_x<<<dim3((N_NODES * KPAD_IN + 255) / 256), dim3(256), 0, stream>>>(x, xb);
    k_cvt_wt<<<dim3((HID * KPAD_IN + 255) / 256), dim3(256), 0, stream>>>(W_in, Wint, F_IN, HID, KPAD_IN);
    for (int i = 0; i < NLAYER; ++i) {
        k_cvt_wt<<<dim3((HID * HID + 255) / 256), dim3(256), 0, stream>>>(
            Wl + (size_t)i * HID * HID, Wlt + (size_t)i * HID * HID, HID, HID, HID);
        k_cvt_wt<<<dim3((HID * HID + 255) / 256), dim3(256), 0, stream>>>(
            Wr + (size_t)i * HID * HID, Wrt + (size_t)i * HID * HID, HID, HID, HID);
    }
    k_cvt_wt<<<dim3((HID * HID + 255) / 256), dim3(256), 0, stream>>>(Wq, Wqt, HID, HID, HID);
    k_cvt_wt<<<dim3((HID * HID + 255) / 256), dim3(256), 0, stream>>>(Wk, Wkt, HID, HID, HID);
    k_cvt_wt<<<dim3((HID * HID + 255) / 256), dim3(256), 0, stream>>>(Wv, Wvt, HID, HID, HID);
    k_cvt_wt<<<dim3((HID * HID + 255) / 256), dim3(256), 0, stream>>>(Wskip, Wst, HID, HID, HID);

    // --- input layer: h = relu(bn(x @ W_in + b_in)) ---
    k_gemm_bf16<KPAD_IN><<<dim3((N_NODES + 63) / 64), dim3(128), 0, stream>>>(
        xb, Wint, b_in, bufA, N_NODES);
    bn(bufA, nullptr, g_in, be_in, nullptr, h, hb, 1);

    // --- GATv2 layers ---
    const int E2 = N_EDGES + N_NODES;               // with self loops
    for (int i = 0; i < NLAYER; ++i) {
        gemm128(hb, Wlt + (size_t)i * HID * HID, bl + (size_t)i * HID, bufA); // xl
        gemm128(hb, Wrt + (size_t)i * HID * HID, br + (size_t)i * HID, bufB); // xr
        fill(nmax, -INFINITY, N_NODES * NHEAD);
        fill(nsum, 0.0f, N_NODES * NHEAD);
        fill(bufC, 0.0f, N_NODES * HID);
        int tl = E2 * NHEAD;
        k_gat_logit<<<dim3((tl + 255) / 256), dim3(256), 0, stream>>>(
            bufA, bufB, ei, att + (size_t)i * NHEAD * CH, elog, nmax, tl);
        k_expsum<<<dim3((tl + 255) / 256), dim3(256), 0, stream>>>(
            elog, ei, nmax, nsum, tl, 1);
        int ts = E2 * HID;
        k_scatter<<<dim3((ts + 255) / 256), dim3(256), 0, stream>>>(
            bufA, elog, nsum, ei, bufC, ts, 1);
        bn(bufC, gbias + (size_t)i * HID, g_gat + (size_t)i * HID,
           be_gat + (size_t)i * HID, h, h, hb, 3);  // h += relu(bn(out + gbias))
    }

    // --- TransformerConv ---
    gemm128(hb, Wqt, bq, bufA);                     // q
    gemm128(hb, Wkt, bk, bufB);                     // k
    gemm128(hb, Wvt, bv, bufC);                     // v
    gemm128(hb, Wst, bskip, bufD);                  // tout starts as skip
    fill(nmax, -INFINITY, N_NODES * NHEAD);
    fill(nsum, 0.0f, N_NODES * NHEAD);
    {
        int tl = N_EDGES * NHEAD;
        k_tr_logit<<<dim3((tl + 255) / 256), dim3(256), 0, stream>>>(
            bufA, bufB, ei, elog, nmax, tl);
        k_expsum<<<dim3((tl + 255) / 256), dim3(256), 0, stream>>>(
            elog, ei, nmax, nsum, tl, 0);
        int ts = N_EDGES * HID;
        k_scatter<<<dim3((ts + 255) / 256), dim3(256), 0, stream>>>(
            bufC, elog, nsum, ei, bufD, ts, 0);
    }
    bn(bufD, nullptr, g_tr, be_tr, h, h, nullptr, 2);   // h += bn(tout)

    // --- pooling ---
    fill(psum, 0.0f, N_GRAPHS * HID);
    fill(pmax, -INFINITY, N_GRAPHS * HID);
    fill(cnt, 0.0f, N_GRAPHS);
    k_count<<<dim3((N_NODES + 255) / 256), dim3(256), 0, stream>>>(batch, cnt);
    k_pool_acc<<<dim3((N_NODES * HID + 255) / 256), dim3(256), 0, stream>>>(h, batch, psum, pmax);
    k_pool_final<<<dim3((N_GRAPHS * HID + 255) / 256), dim3(256), 0, stream>>>(psum, pmax, cnt, out);
}
